// AMSCF_52158082843320
// MI455X (gfx1250) — compile-verified, matrix operand order checked by reading, not run on hardware
//
#include <hip/hip_runtime.h>
#include <hip/hip_bf16.h>

// ---------------------------------------------------------------------------
// AMSCF forward for MI455X (gfx1250, wave32). All GEMMs run on
// v_wmma_f32_16x16x32_bf16 with fully vectorized b128 fragment loads;
// large intermediates stored bf16 to halve HBM traffic (workload is
// bandwidth-bound at 23.3 TB/s, ~315 GFLOP total).
// ---------------------------------------------------------------------------

typedef __attribute__((ext_vector_type(16))) __bf16 v16bf;
typedef __attribute__((ext_vector_type(8)))  float  v8f;
typedef __attribute__((ext_vector_type(4)))  unsigned int u32x4;

union Frag16 {
    v16bf v;
    unsigned short u[16];
    u32x4 q[2];
};

__device__ __forceinline__ unsigned short f2bf(float f) {
    unsigned int u = __float_as_uint(f);
    u += 0x7FFFu + ((u >> 16) & 1u);          // RNE
    return (unsigned short)(u >> 16);
}
__device__ __forceinline__ float bf2f(unsigned short s) {
    return __uint_as_float(((unsigned int)s) << 16);
}
__device__ __forceinline__ float sigmoidf_(float x) { return 1.0f / (1.0f + expf(-x)); }

// ---------------------------------------------------------------------------
// bf16 WMMA GEMM: one wave (32 threads) computes a 16x32 f32 tile (two
// adjacent 16x16 n-tiles sharing one A fragment -> 2 WMMAs per 3 fragment
// loads). All operands are K-contiguous:
//   A(row, k) = A[row*a_row_stride + a_off + k]
//   B(k,col)  = B[col*ldb + k]            (i.e. B^T row-major)
// K must be a multiple of 32; gridDim = (Mtiles, Ntiles/2, batch).
// ---------------------------------------------------------------------------
template <bool OUT_BF16>
__global__ void gemm_wmma(const unsigned short* __restrict__ A, long a_row_stride, long a_off,
                          const unsigned short* __restrict__ Bp, long ldb, long b_batch_stride,
                          void* __restrict__ Out, long out_row_stride, long out_batch_stride,
                          const float* __restrict__ bias, int Mreal, int K)
{
    const int mtile = blockIdx.x;
    const int npair = blockIdx.y;
    const int batch = blockIdx.z;
    const int lane  = threadIdx.x & 31;
    const int half  = lane >> 4;     // 0 or 1
    const int lm    = lane & 15;

    const unsigned short* Bb = Bp + (long)batch * b_batch_stride;
    const long arow  = (long)(mtile * 16 + lm) * a_row_stride + a_off;
    const long brow0 = (long)(npair * 32 + lm) * ldb;
    const long brow1 = brow0 + 16 * ldb;

    v8f acc0 = {};
    v8f acc1 = {};
    for (int k = 0; k < K; k += 32) {
        // speculative prefetch four k-groups (128 elements) ahead (global_prefetch_b8;
        // invalid addresses are silently dropped per CDNA5 ISA)
        __builtin_prefetch(A  + arow  + k + 128, 0, 1);
        __builtin_prefetch(Bb + brow0 + k + 128, 0, 1);
        __builtin_prefetch(Bb + brow1 + k + 128, 0, 1);

        Frag16 af, bf0, bf1;
        // ISA 16x32 bf16 fragment: lane holds two contiguous 8x bf16 runs
        af.q[0]  = *(const u32x4*)(A  + arow  + k + half * 8);
        af.q[1]  = *(const u32x4*)(A  + arow  + k + 16 + half * 8);
        bf0.q[0] = *(const u32x4*)(Bb + brow0 + k + half * 8);
        bf0.q[1] = *(const u32x4*)(Bb + brow0 + k + 16 + half * 8);
        bf1.q[0] = *(const u32x4*)(Bb + brow1 + k + half * 8);
        bf1.q[1] = *(const u32x4*)(Bb + brow1 + k + 16 + half * 8);

        acc0 = __builtin_amdgcn_wmma_f32_16x16x32_bf16(false, af.v, false, bf0.v,
                                                       (short)0, acc0, false, false);
        acc1 = __builtin_amdgcn_wmma_f32_16x16x32_bf16(false, af.v, false, bf1.v,
                                                       (short)0, acc1, false, false);
    }

    const int col0 = npair * 32 + lm;
    const int col1 = col0 + 16;
    const float bv0 = bias ? bias[col0] : 0.0f;
    const float bv1 = bias ? bias[col1] : 0.0f;
#pragma unroll
    for (int r = 0; r < 8; ++r) {
        const int row = mtile * 16 + r + 8 * half;   // D layout: VGPR r, lane half
        if (row < Mreal) {
            const long base = (long)batch * out_batch_stride + (long)row * out_row_stride;
            const float v0 = acc0[r] + bv0;
            const float v1 = acc1[r] + bv1;
            if (OUT_BF16) {
                ((unsigned short*)Out)[base + col0] = f2bf(v0);
                ((unsigned short*)Out)[base + col1] = f2bf(v1);
            } else {
                ((float*)Out)[base + col0] = v0;
                ((float*)Out)[base + col1] = v1;
            }
        }
    }
}

// ---------------------------------------------------------------------------
// Graph preparation kernels
// ---------------------------------------------------------------------------
__global__ void embed_scores(const float* __restrict__ E1, const float* __restrict__ E2,
                             float* __restrict__ R, int total)
{
    int idx = blockIdx.x * 256 + threadIdx.x;
    if (idx >= total) return;
    int n = idx / 1000, m = idx % 1000;
    float s = 0.f;
#pragma unroll
    for (int d = 0; d < 10; ++d) s += E1[n * 10 + d] * E2[m * 10 + d];
    R[idx] = fmaxf(s, 0.f);
}

__global__ void softmax_graphs(const float* __restrict__ R, const float* __restrict__ A,
                               const float* __restrict__ FSP, const float* __restrict__ DSP,
                               const float* __restrict__ beta1, const float* __restrict__ beta2,
                               float* __restrict__ IA, float* __restrict__ C1, float* __restrict__ C2)
{
    const int n = blockIdx.x, tid = threadIdx.x;
    __shared__ float red[256];
    float mx = -1e30f;
    for (int m = tid; m < 1000; m += 256) mx = fmaxf(mx, R[(long)n * 1000 + m]);
    red[tid] = mx; __syncthreads();
    for (int st = 128; st > 0; st >>= 1) { if (tid < st) red[tid] = fmaxf(red[tid], red[tid + st]); __syncthreads(); }
    mx = red[0]; __syncthreads();
    float s = 0.f;
    for (int m = tid; m < 1000; m += 256) s += expf(R[(long)n * 1000 + m] - mx);
    red[tid] = s; __syncthreads();
    for (int st = 128; st > 0; st >>= 1) { if (tid < st) red[tid] += red[tid + st]; __syncthreads(); }
    s = red[0]; __syncthreads();
    const float w1 = sigmoidf_(beta1[0]);
    const float w2 = sigmoidf_(beta2[0]);
    for (int m = tid; m < 1000; m += 256) {
        long i = (long)n * 1000 + m;
        float aadp = expf(R[i] - mx) / s;
        float av = A[i];
        IA[i] = 0.5f * av + 0.5f * aadp;
        C1[i] = w1 * av + (1.f - w1) * FSP[i];
        C2[i] = w2 * av + (1.f - w2) * DSP[i];
    }
}

__global__ void fusion_partial(const float* __restrict__ IA, const float* __restrict__ C1,
                               const float* __restrict__ C2, const float* __restrict__ fW,
                               const float* __restrict__ fb, const float* __restrict__ fv,
                               float* __restrict__ part)
{
    const int k = blockIdx.x / 1000;
    const int n = blockIdx.x % 1000;
    const float* G = (k == 0) ? IA : (k == 1) ? C1 : C2;
    __shared__ float row[1000];
    __shared__ float acc[32];
    const int tid = threadIdx.x;  // 128
    for (int m = tid; m < 1000; m += 128) row[m] = G[(long)n * 1000 + m];
    __syncthreads();
    if (tid < 32) {
        float s = fb[tid];
        for (int m = 0; m < 1000; ++m) s += row[m] * fW[tid * 1000 + m];
        acc[tid] = tanhf(s) * fv[tid];
    }
    __syncthreads();
    if (tid == 0) {
        float s = 0.f;
        for (int a = 0; a < 32; ++a) s += acc[a];
        part[k * 1000 + n] = s;
    }
}

__global__ void reduce_alpha(const float* __restrict__ part, float* __restrict__ alpha)
{
    __shared__ float red[256];
    __shared__ float scs[3];
    const int tid = threadIdx.x;
    for (int k = 0; k < 3; ++k) {
        float s = 0.f;
        for (int i = tid; i < 1000; i += 256) s += part[k * 1000 + i];
        red[tid] = s; __syncthreads();
        for (int st = 128; st > 0; st >>= 1) { if (tid < st) red[tid] += red[tid + st]; __syncthreads(); }
        if (tid == 0) scs[k] = red[0] / 1000.f;
        __syncthreads();
    }
    if (tid == 0) {
        float mx = fmaxf(scs[0], fmaxf(scs[1], scs[2]));
        float e0 = expf(scs[0] - mx), e1 = expf(scs[1] - mx), e2 = expf(scs[2] - mx);
        float s = e0 + e1 + e2;
        alpha[0] = e0 / s; alpha[1] = e1 / s; alpha[2] = e2 / s;
    }
}

__global__ void build_MG_rowsum(const float* __restrict__ IA, const float* __restrict__ C1,
                                const float* __restrict__ C2, const float* __restrict__ alpha,
                                float* __restrict__ MG, float* __restrict__ dvec)
{
    const int n = blockIdx.x, tid = threadIdx.x;
    const float a0 = alpha[0], a1 = alpha[1], a2 = alpha[2];
    __shared__ float red[256];
    float s = 0.f;
    for (int m = tid; m < 1000; m += 256) {
        long i = (long)n * 1000 + m;
        float v = a0 * IA[i] + a1 * C1[i] + a2 * C2[i];
        if (m == n) v += 1.f;
        MG[i] = v;
        s += v;
    }
    red[tid] = s; __syncthreads();
    for (int st = 128; st > 0; st >>= 1) { if (tid < st) red[tid] += red[tid + st]; __syncthreads(); }
    if (tid == 0) dvec[n] = red[0];
}

__global__ void build_S16(const float* __restrict__ MG, const float* __restrict__ dvec,
                          unsigned short* __restrict__ S16)
{
    int idx = blockIdx.x * 256 + threadIdx.x;
    if (idx >= 1024 * 1024) return;
    int r = idx >> 10, c = idx & 1023;
    float v = 0.f;
    if (r < 1000 && c < 1000) {
        float dr = rsqrtf(fmaxf(dvec[r], 1e-12f));
        float dc = rsqrtf(fmaxf(dvec[c], 1e-12f));
        v = MG[(long)r * 1000 + c] * dr * dc;
    }
    S16[idx] = f2bf(v);
}

// ---------------------------------------------------------------------------
// GCN staging / pointwise kernels
// ---------------------------------------------------------------------------
__global__ void build_xb(const float* __restrict__ x, unsigned short* __restrict__ xb, int total)
{
    int idx = blockIdx.x * 256 + threadIdx.x;
    if (idx >= total) return;
    int bt = idx >> 10, n = idx & 1023;
    float v = (n < 1000) ? x[(long)bt * 1000 + n] : 0.f;
    xb[idx] = f2bf(v);
}

// relu(LayerNorm) of GCN layer 1, written TRANSPOSED: hT[bt][c][n] (n padded
// to 1024) so the heavy S@h GEMM gets K-contiguous b128 B-fragment loads.
__global__ void gcn1_ln_relu(const float* __restrict__ u, const float* __restrict__ W1,
                             const float* __restrict__ b1, const float* __restrict__ g,
                             const float* __restrict__ bb, unsigned short* __restrict__ h16T)
{
    const int bt = blockIdx.x / 1000;
    const int m  = blockIdx.x % 1000;
    const int c  = threadIdx.x;  // 96
    __shared__ float sh[96];
    __shared__ float st[2];
    const float uv = u[m * 384 + bt];
    const float hv = uv * W1[c] + b1[c];
    sh[c] = hv;
    __syncthreads();
    if (c == 0) {
        float s = 0.f, s2 = 0.f;
        for (int i = 0; i < 96; ++i) { s += sh[i]; s2 += sh[i] * sh[i]; }
        float mean = s / 96.f;
        st[0] = mean;
        st[1] = rsqrtf(s2 / 96.f - mean * mean + 1e-5f);
    }
    __syncthreads();
    float y = (hv - st[0]) * st[1] * g[c] + bb[c];
    y = fmaxf(y, 0.f);
    h16T[(long)bt * 98304 + c * 1024 + m] = f2bf(y);
}

__global__ void zero_h16_pad(unsigned short* __restrict__ h16T)
{
    int idx = blockIdx.x * 256 + threadIdx.x;          // 384 * 96 * 24 total
    if (idx >= 384 * 96 * 24) return;
    int bt  = idx / (96 * 24);
    int rem = idx % (96 * 24);
    int c   = rem / 24;
    int m   = 1000 + rem % 24;
    h16T[(long)bt * 98304 + c * 1024 + m] = 0;
}

__global__ void ln2_kernel(const unsigned short* __restrict__ y, const float* __restrict__ g,
                           const float* __restrict__ b, unsigned short* __restrict__ o)
{
    const int row = blockIdx.x, c = threadIdx.x;  // 96
    __shared__ float sh[96];
    __shared__ float st[2];
    const float v = bf2f(y[(long)row * 96 + c]);
    sh[c] = v;
    __syncthreads();
    if (c == 0) {
        float s = 0.f, s2 = 0.f;
        for (int i = 0; i < 96; ++i) { s += sh[i]; s2 += sh[i] * sh[i]; }
        float mean = s / 96.f;
        st[0] = mean;
        st[1] = rsqrtf(s2 / 96.f - mean * mean + 1e-5f);
    }
    __syncthreads();
    o[(long)row * 96 + c] = f2bf((v - st[0]) * st[1] * g[c] + b[c]);
}

__global__ void cvt_bf16(const float* __restrict__ s, unsigned short* __restrict__ d, int n)
{
    int idx = blockIdx.x * 256 + threadIdx.x;
    if (idx < n) d[idx] = f2bf(s[idx]);
}

__global__ void cvt_bf16_T(const float* __restrict__ s, unsigned short* __restrict__ d, int R, int C)
{
    int idx = blockIdx.x * 256 + threadIdx.x;
    if (idx >= R * C) return;
    int r = idx / C, c = idx % C;
    d[(long)c * R + r] = f2bf(s[idx]);
}

// ---------------------------------------------------------------------------
// GRU kernels
// ---------------------------------------------------------------------------
__global__ void zero_hs(float* __restrict__ hs, unsigned short* __restrict__ hs16, int n)
{
    int idx = blockIdx.x * 256 + threadIdx.x;
    if (idx < n) { hs[idx] = 0.f; hs16[idx] = 0; }
}

__global__ void gru_gates(const float* __restrict__ xp, const float* __restrict__ hp,
                          float* __restrict__ hs, unsigned short* __restrict__ hs16,
                          unsigned short* __restrict__ oseq, int t, int total)
{
    int idx = blockIdx.x * 256 + threadIdx.x;
    if (idx >= total) return;
    const int bn = idx / 192, c = idx % 192;
    const long b3 = (long)bn * 576;
    const float r = sigmoidf_(xp[b3 + c]       + hp[b3 + c]);
    const float z = sigmoidf_(xp[b3 + 192 + c] + hp[b3 + 192 + c]);
    const float n = tanhf(xp[b3 + 384 + c] + r * hp[b3 + 384 + c]);
    const float h = hs[(long)bn * 192 + c];
    const float hn = (1.f - z) * n + z * h;
    hs[(long)bn * 192 + c]   = hn;
    hs16[(long)bn * 192 + c] = f2bf(hn);
    oseq[(long)bn * 2304 + t * 192 + c] = f2bf(hn);
}

// ---------------------------------------------------------------------------
// Attention pooling + FC head
// ---------------------------------------------------------------------------
__global__ void attn_fc(const unsigned short* __restrict__ out2, const float* __restrict__ att_w,
                        const float* __restrict__ fc1W, const float* __restrict__ fc1b,
                        const float* __restrict__ fc2W, const float* __restrict__ fc2b,
                        float* __restrict__ out)
{
    const int bn = blockIdx.x, tid = threadIdx.x;  // 192
    __shared__ float o[2304];
    __shared__ float att[12];
    __shared__ float ctx[192];
    __shared__ float hfc[192];
    for (int i = tid; i < 2304; i += 192) o[i] = bf2f(out2[(long)bn * 2304 + i]);
    __syncthreads();
    if (tid < 12) {
        float s = 0.f;
        for (int c = 0; c < 192; ++c) s += o[tid * 192 + c] * att_w[c];
        att[tid] = s;
    }
    __syncthreads();
    if (tid == 0) {
        float mx = att[0];
        for (int i = 1; i < 12; ++i) mx = fmaxf(mx, att[i]);
        float s = 0.f;
        for (int i = 0; i < 12; ++i) { att[i] = expf(att[i] - mx); s += att[i]; }
        for (int i = 0; i < 12; ++i) att[i] /= s;
    }
    __syncthreads();
    {
        float s = 0.f;
        for (int tt = 0; tt < 12; ++tt) s += att[tt] * o[tt * 192 + tid];
        ctx[tid] = s;
    }
    __syncthreads();
    {
        float a = fc1b[tid];
        for (int k = 0; k < 192; ++k) a += ctx[k] * fc1W[tid * 192 + k];
        hfc[tid] = 0.5f * a * (1.f + tanhf(0.7978845608028654f * (a + 0.044715f * a * a * a)));
    }
    __syncthreads();
    if (tid < 12) {
        float dv = fc2b[tid];
        for (int k = 0; k < 192; ++k) dv += hfc[k] * fc2W[tid * 192 + k];
        int b = bn / 1000, n = bn % 1000;
        out[(long)b * 12000 + tid * 1000 + n] = dv;
    }
}

// ---------------------------------------------------------------------------
// Orchestration
// ---------------------------------------------------------------------------
extern "C" void kernel_launch(void* const* d_in, const int* in_sizes, int n_in,
                              void* d_out, int out_size, void* d_ws, size_t ws_size,
                              hipStream_t stream)
{
    const float* x    = (const float*)d_in[0];
    const float* A    = (const float*)d_in[1];
    const float* FSP  = (const float*)d_in[2];
    const float* DSP  = (const float*)d_in[3];
    const float* E1   = (const float*)d_in[4];
    const float* E2   = (const float*)d_in[5];
    const float* fusW = (const float*)d_in[6];
    const float* fusb = (const float*)d_in[7];
    const float* fusv = (const float*)d_in[8];
    const float* bt1  = (const float*)d_in[9];
    const float* bt2  = (const float*)d_in[10];
    const float* W1   = (const float*)d_in[11];
    const float* b1   = (const float*)d_in[12];
    const float* ln1g = (const float*)d_in[13];
    const float* ln1b = (const float*)d_in[14];
    const float* W2   = (const float*)d_in[15];
    const float* b2   = (const float*)d_in[16];
    const float* ln2g = (const float*)d_in[17];
    const float* ln2b = (const float*)d_in[18];
    const float* Wih0 = (const float*)d_in[19];
    const float* Whh0 = (const float*)d_in[20];
    const float* bih0 = (const float*)d_in[21];
    const float* bhh0 = (const float*)d_in[22];
    const float* Wih1 = (const float*)d_in[23];
    const float* Whh1 = (const float*)d_in[24];
    const float* bih1 = (const float*)d_in[25];
    const float* bhh1 = (const float*)d_in[26];
    const float* attw = (const float*)d_in[27];
    const float* fc1W = (const float*)d_in[28];
    const float* fc1b = (const float*)d_in[29];
    const float* fc2W = (const float*)d_in[30];
    const float* fc2b = (const float*)d_in[31];
    float* out = (float*)d_out;

    const size_t MB = 1ull << 20;
    if (ws_size < 566 * MB) return;  // static workspace layout below
    char* w = (char*)d_ws;

    float* R_    = (float*)(w + 0);                 // 4MB  (scores, reused as MG)
    float* IA    = (float*)(w + 4 * MB);            // 4MB
    float* C1    = (float*)(w + 8 * MB);            // 4MB
    float* C2    = (float*)(w + 12 * MB);           // 4MB
    float* dvec  = (float*)(w + 16 * MB);           // 4KB
    float* alpha = (float*)(w + 16 * MB + 8192);    // 12B
    float* part  = (float*)(w + 16 * MB + 16384);   // 12KB
    unsigned short* S16   = (unsigned short*)(w + 17 * MB);   // 2MB (1024x1024)
    unsigned short* xb    = (unsigned short*)(w + 19 * MB);   // 768KB (384x1024)
    float*          u     = (float*)(w + 20 * MB);            // 1.5MB (1000x384)
    unsigned short* w2bT  = (unsigned short*)(w + 22 * MB);
    unsigned short* wih0b = (unsigned short*)(w + 22 * MB + 64 * 1024);
    unsigned short* whh0b = (unsigned short*)(w + 22 * MB + 256 * 1024);
    unsigned short* wih1b = (unsigned short*)(w + 22 * MB + 512 * 1024);
    unsigned short* whh1b = (unsigned short*)(w + 22 * MB + 768 * 1024);
    unsigned short* h16T = (unsigned short*)(w + 24 * MB);    // 75.5MB (384x96x1024)
    float*          hp   = (float*)(w + 24 * MB);             // reuse of h16T after big GEMM
    unsigned short* g1   = (unsigned short*)(w + 100 * MB);   // 73.7MB (384000x96)
    unsigned short* xseq = (unsigned short*)(w + 100 * MB);   // reuse of g1 after W2 GEMM
    unsigned short* y2b  = (unsigned short*)(w + 172 * MB);   // 73.7MB
    float*          xp   = (float*)(w + 172 * MB);            // reuse of y2b after LN2
    float*          hs   = (float*)(w + 244 * MB);            // 24.6MB (32000x192)
    unsigned short* hs16 = (unsigned short*)(w + 268 * MB);   // 12.3MB
    unsigned short* out1 = (unsigned short*)(w + 280 * MB);   // 147.5MB (32000x12x192)
    unsigned short* out2 = (unsigned short*)(w + 424 * MB);   // 147.5MB

    const dim3 b256(256);

    // --- graph prep ---
    embed_scores<<<dim3((1000000 + 255) / 256), b256, 0, stream>>>(E1, E2, R_, 1000000);
    softmax_graphs<<<dim3(1000), b256, 0, stream>>>(R_, A, FSP, DSP, bt1, bt2, IA, C1, C2);
    fusion_partial<<<dim3(3000), dim3(128), 0, stream>>>(IA, C1, C2, fusW, fusb, fusv, part);
    reduce_alpha<<<dim3(1), b256, 0, stream>>>(part, alpha);
    build_MG_rowsum<<<dim3(1000), b256, 0, stream>>>(IA, C1, C2, alpha, R_, dvec);
    build_S16<<<dim3((1024 * 1024) / 256), b256, 0, stream>>>(R_, dvec, S16);

    // --- GCN layer 1: u = S @ X  (M=1008/63 tiles, N=384 -> 12 pairs, K=1024) ---
    build_xb<<<dim3((384 * 1024) / 256), b256, 0, stream>>>(x, xb, 384 * 1024);
    gemm_wmma<false><<<dim3(63, 12, 1), dim3(32), 0, stream>>>(
        S16, 1024, 0, xb, 1024, 0, u, 384, 0, nullptr, 1000, 1024);
    gcn1_ln_relu<<<dim3(384000), dim3(96), 0, stream>>>(u, W1, b1, ln1g, ln1b, h16T);
    zero_h16_pad<<<dim3((384 * 96 * 24 + 255) / 256), b256, 0, stream>>>(h16T);

    // --- weight bf16 staging ---
    cvt_bf16_T<<<dim3((96 * 96 + 255) / 256), b256, 0, stream>>>(W2, w2bT, 96, 96);
    cvt_bf16<<<dim3((576 * 96 + 255) / 256), b256, 0, stream>>>(Wih0, wih0b, 576 * 96);
    cvt_bf16<<<dim3((576 * 192 + 255) / 256), b256, 0, stream>>>(Whh0, whh0b, 576 * 192);
    cvt_bf16<<<dim3((576 * 192 + 255) / 256), b256, 0, stream>>>(Wih1, wih1b, 576 * 192);
    cvt_bf16<<<dim3((576 * 192 + 255) / 256), b256, 0, stream>>>(Whh1, whh1b, 576 * 192);

    // --- GCN layer 2a (heavy GEMM): g1[bt] = S @ h[bt], batched over bt=384.
    //     B = hT[bt][c][n] is K-contiguous -> b128 fragment loads. N=96 -> 3 pairs.
    gemm_wmma<true><<<dim3(63, 3, 384), dim3(32), 0, stream>>>(
        S16, 1024, 0, h16T, 1024, 98304, g1, 96, 96000, nullptr, 1000, 1024);

    // --- GCN layer 2b: y2 = g1 @ W2 + b2  (M=384000, N=96 -> 3 pairs, K=96) ---
    gemm_wmma<true><<<dim3(24000, 3, 1), dim3(32), 0, stream>>>(
        g1, 96, 0, w2bT, 96, 0, y2b, 96, 0, b2, 384000, 96);

    // --- LN2 -> GRU input sequence (flat reinterpret matches reference view) ---
    ln2_kernel<<<dim3(384000), dim3(96), 0, stream>>>(y2b, ln2g, ln2b, xseq);

    // --- two GRU layers, T=12 scan; per step: x-proj GEMM, h-proj GEMM, gates ---
    for (int layer = 0; layer < 2; ++layer) {
        zero_hs<<<dim3((6144000 + 255) / 256), b256, 0, stream>>>(hs, hs16, 6144000);
        const unsigned short* inseq = (layer == 0) ? xseq : out1;
        const long ars = (layer == 0) ? 1152 : 2304;    // T*K row stride
        const int  K   = (layer == 0) ? 96 : 192;
        const unsigned short* wih = (layer == 0) ? wih0b : wih1b;
        const unsigned short* whh = (layer == 0) ? whh0b : whh1b;
        const float* bih = (layer == 0) ? bih0 : bih1;
        const float* bhh = (layer == 0) ? bhh0 : bhh1;
        unsigned short* oseq = (layer == 0) ? out1 : out2;
        for (int t = 0; t < 12; ++t) {
            gemm_wmma<false><<<dim3(2000, 18, 1), dim3(32), 0, stream>>>(
                inseq, ars, (long)t * K, wih, K, 0, xp, 576, 0, bih, 32000, K);
            gemm_wmma<false><<<dim3(2000, 18, 1), dim3(32), 0, stream>>>(
                hs16, 192, 0, whh, 192, 0, hp, 576, 0, bhh, 32000, 192);
            gru_gates<<<dim3((6144000 + 255) / 256), b256, 0, stream>>>(
                xp, hp, hs, hs16, oseq, t, 6144000);
        }
    }

    // --- attention pooling + FC head -> (B, Hor, N) output ---
    attn_fc<<<dim3(32000), dim3(192), 0, stream>>>(out2, attw, fc1W, fc1b, fc2W, fc2b, out);
}
